// RBF_75797582840427
// MI455X (gfx1250) — compile-verified, matrix operand order checked by reading, not run on hardware
//
#include <hip/hip_runtime.h>
#include <math.h>

// ---------------------------------------------------------------------------
// MI455X (gfx1250) fused RBF + MLP head.
//   out = ReLU(BN2(ReLU(BN1(cdist(x,c)*sigma)) @ W1^T + b1)) @ W2^T + b2
// GEMMs: V_WMMA_F32_16X16X32_BF16 (wave32, fp32 accum), operands staged into
// double-buffered LDS by the Tensor Data Mover (tensor_load_to_lds, TENSORcnt)
// so matrix math overlaps DMA.
// Block = 256 threads = 8 waves (4M x 2N), block tile 64M x 128N, K chunk 64.
// ---------------------------------------------------------------------------

typedef __bf16 bf16_t;
typedef bf16_t v16bf __attribute__((ext_vector_type(16)));
typedef bf16_t v8bf  __attribute__((ext_vector_type(8)));
typedef float  v8f   __attribute__((ext_vector_type(8)));
typedef unsigned int u32x4 __attribute__((ext_vector_type(4)));
typedef int i32x8 __attribute__((ext_vector_type(8)));
typedef int i32x4 __attribute__((ext_vector_type(4)));

#define BN_EPS 1e-5f
#define WPB 8            // waves per block
#define LDS_STRIDE 72    // 64 bf16 cols + 8 bf16 TDM pad (row stride 144B)

static constexpr int Bn  = 512;   // batch
static constexpr int Cn  = 2408;  // centers
static constexpr int Fn  = 875;   // features
static constexpr int Hn  = 1024;  // hidden
static constexpr int On  = 206;   // output
static constexpr int FP  = 896;   // F padded (14*64)
static constexpr int CP  = 2432;  // C padded (19*128 as N, 38*64 as K)
static constexpr int OPD = 256;   // O padded (2*128)

__device__ __forceinline__ unsigned short f32_to_bf16(float f) {
  unsigned u = __float_as_uint(f);
  u += 0x7FFFu + ((u >> 16) & 1u);          // round-to-nearest-even
  return (unsigned short)(u >> 16);
}

__device__ __forceinline__ unsigned lds_off_of(const void* p) {
  // generic pointers to LDS carry the allocation-relative offset in addr[31:0]
  return (unsigned)(unsigned long long)(uintptr_t)p;
}

// ---- Tensor Data Mover: 2D tile (tile_w x tile_h, 2-byte elems) -> LDS -----
// D# per CDNA5 ISA 8.3-8.5. pad_enable: interval=32 DWORDs (one 64-elem row),
// amount=4 DWORDs (16B) -> LDS row stride 144B (16B aligned, bank-conflict
// free for 16 consecutive rows).
__device__ __forceinline__ void tdm_load_2d(unsigned lds_addr, const bf16_t* gsrc,
                                            unsigned tensor_w, unsigned tensor_h,
                                            unsigned stride_elems,
                                            unsigned tile_w, unsigned tile_h) {
  unsigned long long ga = (unsigned long long)(uintptr_t)gsrc;
  u32x4 g0;
  g0[0] = 1u;                                   // count=1, user descriptor
  g0[1] = lds_addr;                             // lds_addr[31:0]
  g0[2] = (unsigned)ga;                         // global_addr[31:0]
  g0[3] = (unsigned)(ga >> 32) | 0x80000000u;   // global_addr[56:32] | type=2
  i32x8 g1;
  g1[0] = (int)((1u << 16)      // data_size = 1 -> 2 bytes
              | (1u << 20)      // pad_enable
              | (4u << 22)      // pad_interval: 32 DWORDs (one row)
              | (3u << 25));    // pad_amount: 4 DWORDs (16B)
  g1[1] = (int)((tensor_w & 0xFFFFu) << 16);                      // dim0[15:0]
  g1[2] = (int)((tensor_w >> 16) | ((tensor_h & 0xFFFFu) << 16)); // dim0 hi | dim1 lo
  g1[3] = (int)((tensor_h >> 16) | (tile_w << 16));               // dim1 hi | tile_dim0
  g1[4] = (int)(tile_h & 0xFFFFu);                                // tile_dim1 (tile_dim2=0)
  g1[5] = (int)stride_elems;                                      // dim0_stride[31:0]
  g1[6] = 0;                                                      // dim0_stride hi | dim1_stride lo
  g1[7] = 0;
  i32x4 gz4 = {0, 0, 0, 0};
  i32x8 gz8 = {0, 0, 0, 0, 0, 0, 0, 0};
  // clang-23 / amdgpu-toolchain 6-arg form: (g0, g1, g2, g3, extra, cpol)
  __builtin_amdgcn_tensor_load_to_lds(g0, g1, gz4, gz4, gz8, 0);
}

// per-lane WMMA 16-bit fragment from an LDS row (A 16x32 layout, B mirrored):
// elems 0..7 <- K = k0..k0+7 ; elems 8..15 <- K = k0+16..k0+23 ; k0 = kin+hi*8
__device__ __forceinline__ v16bf lds_frag(const bf16_t* __restrict__ row, int k0) {
  union { v16bf v; v8bf h[2]; } u;
  u.h[0] = *reinterpret_cast<const v8bf*>(row + k0);
  u.h[1] = *reinterpret_cast<const v8bf*>(row + k0 + 16);
  return u.v;
}

// one 64-deep K chunk of WMMAs for this wave's 16x64 tile
__device__ __forceinline__ void mma_chunk(const bf16_t* __restrict__ sA,
                                          const bf16_t* __restrict__ sB,
                                          int wm, int wn, int lrow, int hi,
                                          v8f& a0, v8f& a1, v8f& a2, v8f& a3) {
  const bf16_t* ar  = sA + (wm * 16 + lrow) * LDS_STRIDE;
  const bf16_t* br0 = sB + (wn * 64 + lrow) * LDS_STRIDE;
#pragma unroll
  for (int kin = 0; kin < 64; kin += 32) {
    const int k0 = kin + hi * 8;
    v16bf av = lds_frag(ar, k0);
    v16bf b0 = lds_frag(br0, k0);
    v16bf b1 = lds_frag(br0 + 16 * LDS_STRIDE, k0);
    v16bf b2 = lds_frag(br0 + 32 * LDS_STRIDE, k0);
    v16bf b3 = lds_frag(br0 + 48 * LDS_STRIDE, k0);
    a0 = __builtin_amdgcn_wmma_f32_16x16x32_bf16(false, av, false, b0, (short)0, a0, false, false);
    a1 = __builtin_amdgcn_wmma_f32_16x16x32_bf16(false, av, false, b1, (short)0, a1, false, false);
    a2 = __builtin_amdgcn_wmma_f32_16x16x32_bf16(false, av, false, b2, (short)0, a2, false, false);
    a3 = __builtin_amdgcn_wmma_f32_16x16x32_bf16(false, av, false, b3, (short)0, a3, false, false);
  }
}

// TDM-fed, double-buffered 64x128 block-tile GEMM core.
#define GEMM_CORE(Abase, Bbase, LDA, ROWS_A, ROWS_B, NK)                        \
  __shared__ __align__(16) bf16_t shA[2][64 * LDS_STRIDE];                      \
  __shared__ __align__(16) bf16_t shB[2][128 * LDS_STRIDE];                     \
  const int lane = threadIdx.x & 31, wave = threadIdx.x >> 5;                   \
  const int wm = wave >> 1, wn = wave & 1;                                      \
  const int lrow = lane & 15, hi = lane >> 4;                                   \
  const int mB = blockIdx.y, nB = blockIdx.x;                                   \
  v8f a0 = {}, a1 = {}, a2 = {}, a3 = {};                                       \
  if (wave == 0) {                                                              \
    tdm_load_2d(lds_off_of(&shA[0][0]), (Abase) + (size_t)mB * 64 * (LDA),      \
                (LDA), (ROWS_A), (LDA), 64, 64);                                \
    tdm_load_2d(lds_off_of(&shB[0][0]), (Bbase) + (size_t)nB * 128 * (LDA),     \
                (LDA), (ROWS_B), (LDA), 64, 128);                               \
  }                                                                             \
  for (int kc = 0; kc < (NK); ++kc) {                                           \
    const int cur = kc & 1;                                                     \
    if (wave == 0) {                                                            \
      if (kc + 1 < (NK)) {                                                      \
        const size_t koff = (size_t)(kc + 1) * 64;                              \
        tdm_load_2d(lds_off_of(&shA[1 - cur][0]),                               \
                    (Abase) + (size_t)mB * 64 * (LDA) + koff,                   \
                    (LDA), (ROWS_A), (LDA), 64, 64);                            \
        tdm_load_2d(lds_off_of(&shB[1 - cur][0]),                               \
                    (Bbase) + (size_t)nB * 128 * (LDA) + koff,                  \
                    (LDA), (ROWS_B), (LDA), 64, 128);                           \
        __builtin_amdgcn_s_wait_tensorcnt(2);  /* chunk kc resident */          \
      } else {                                                                  \
        __builtin_amdgcn_s_wait_tensorcnt(0);                                   \
      }                                                                         \
    }                                                                           \
    __syncthreads();                                                            \
    mma_chunk(shA[cur], shB[cur], wm, wn, lrow, hi, a0, a1, a2, a3);            \
    __syncthreads();  /* done reading before TDM overwrites this buffer */      \
  }

// -------------------------- prep kernels -----------------------------------

__global__ __launch_bounds__(256) void k_cvt_pad(const float* __restrict__ src,
                                                 int srows, int scols,
                                                 bf16_t* __restrict__ dst,
                                                 int drows, int dcols) {
  unsigned short* d = reinterpret_cast<unsigned short*>(dst);
  size_t total = (size_t)drows * dcols;
  for (size_t i = (size_t)blockIdx.x * blockDim.x + threadIdx.x; i < total;
       i += (size_t)gridDim.x * blockDim.x) {
    int r = (int)(i / dcols), c = (int)(i % dcols);
    float v = (r < srows && c < scols) ? src[(size_t)r * scols + c] : 0.f;
    d[i] = f32_to_bf16(v);
  }
}

__global__ __launch_bounds__(256) void k_rowsumsq(const float* __restrict__ src,
                                                  int rows, int cols,
                                                  float* __restrict__ out) {
  int lane = threadIdx.x & 31, wave = threadIdx.x >> 5;
  int row = blockIdx.x * WPB + wave;
  if (row >= rows) return;
  const float* p = src + (size_t)row * cols;
  float s = 0.f;
  for (int i = lane; i < cols; i += 32) { float v = p[i]; s += v * v; }
  for (int off = 16; off > 0; off >>= 1) s += __shfl_xor(s, off, 32);
  if (lane == 0) out[row] = s;
}

// -------------------------- GEMM1: RBF + BN1 + ReLU ------------------------
__global__ __launch_bounds__(256) void k_gemm1_rbf(
    const bf16_t* __restrict__ A, const bf16_t* __restrict__ Bm,
    bf16_t* __restrict__ Hout,
    const float* __restrict__ x2, const float* __restrict__ c2,
    const float* __restrict__ sigma,
    const float* __restrict__ g1p, const float* __restrict__ bt1,
    const float* __restrict__ mu1, const float* __restrict__ vr1) {
  GEMM_CORE(A, Bm, FP, Bn, CP, FP / 64)

  unsigned short* outp = reinterpret_cast<unsigned short*>(Hout);
  const int mbase = mB * 64 + wm * 16 + hi * 8;   // C/D layout: M = vgpr + 8*hi
#pragma unroll
  for (int s = 0; s < 4; ++s) {
    v8f acc = (s == 0) ? a0 : (s == 1) ? a1 : (s == 2) ? a2 : a3;
    const int n = nB * 128 + wn * 64 + s * 16 + lrow;
    const bool valid = (n < Cn);
    float c2n = 0.f, sg = 0.f, sc = 0.f, mu = 0.f, bt = 0.f;
    if (valid) {
      c2n = c2[n]; sg = sigma[n];
      sc = g1p[n] * rsqrtf(vr1[n] + BN_EPS);
      mu = mu1[n]; bt = bt1[n];
    }
#pragma unroll
    for (int v = 0; v < 8; ++v) {
      float hv = 0.f;                       // zero-fill K-padding for GEMM2
      if (valid) {
        float d2 = x2[mbase + v] + c2n - 2.f * acc[v];
        float d = sqrtf(fmaxf(d2, 0.f)) * sg;
        hv = fmaxf((d - mu) * sc + bt, 0.f);
      }
      outp[(size_t)(mbase + v) * CP + n] = f32_to_bf16(hv);
    }
  }
}

// -------------------------- GEMM2: + b1, BN2, ReLU -------------------------
__global__ __launch_bounds__(256) void k_gemm2(
    const bf16_t* __restrict__ A, const bf16_t* __restrict__ Bm,
    bf16_t* __restrict__ Hout, const float* __restrict__ bias,
    const float* __restrict__ g2p, const float* __restrict__ bt2,
    const float* __restrict__ mu2, const float* __restrict__ vr2) {
  GEMM_CORE(A, Bm, CP, Bn, Hn, CP / 64)

  unsigned short* outp = reinterpret_cast<unsigned short*>(Hout);
  const int mbase = mB * 64 + wm * 16 + hi * 8;
#pragma unroll
  for (int s = 0; s < 4; ++s) {
    v8f acc = (s == 0) ? a0 : (s == 1) ? a1 : (s == 2) ? a2 : a3;
    const int n = nB * 128 + wn * 64 + s * 16 + lrow;   // < Hn always
    const float sc = g2p[n] * rsqrtf(vr2[n] + BN_EPS);
    const float mu = mu2[n], bt = bt2[n], bs = bias[n];
#pragma unroll
    for (int v = 0; v < 8; ++v) {
      float hv = acc[v] + bs;
      hv = fmaxf((hv - mu) * sc + bt, 0.f);
      outp[(size_t)(mbase + v) * Hn + n] = f32_to_bf16(hv);
    }
  }
}

// -------------------------- GEMM3: + b2 (fp32 out) -------------------------
__global__ __launch_bounds__(256) void k_gemm3(
    const bf16_t* __restrict__ A, const bf16_t* __restrict__ Bm,
    float* __restrict__ Out, const float* __restrict__ bias) {
  GEMM_CORE(A, Bm, Hn, Bn, OPD, Hn / 64)

  const int mbase = mB * 64 + wm * 16 + hi * 8;
#pragma unroll
  for (int s = 0; s < 4; ++s) {
    v8f acc = (s == 0) ? a0 : (s == 1) ? a1 : (s == 2) ? a2 : a3;
    const int n = nB * 128 + wn * 64 + s * 16 + lrow;
    if (n < On) {
      const float bs = bias[n];
#pragma unroll
      for (int v = 0; v < 8; ++v)
        Out[(size_t)(mbase + v) * On + n] = acc[v] + bs;
    }
  }
}

// -------------------------- launch -----------------------------------------

extern "C" void kernel_launch(void* const* d_in, const int* in_sizes, int n_in,
                              void* d_out, int out_size, void* d_ws, size_t ws_size,
                              hipStream_t stream) {
  const float* inp    = (const float*)d_in[0];
  const float* center = (const float*)d_in[1];
  const float* sigma  = (const float*)d_in[2];
  const float* g1     = (const float*)d_in[3];
  const float* bt1    = (const float*)d_in[4];
  const float* mu1    = (const float*)d_in[5];
  const float* vr1    = (const float*)d_in[6];
  const float* w1     = (const float*)d_in[7];
  const float* b1     = (const float*)d_in[8];
  const float* g2     = (const float*)d_in[9];
  const float* bt2    = (const float*)d_in[10];
  const float* mu2    = (const float*)d_in[11];
  const float* vr2    = (const float*)d_in[12];
  const float* w2     = (const float*)d_in[13];
  const float* b2     = (const float*)d_in[14];
  float* out = (float*)d_out;

  // workspace carve (~14.4 MB)
  char* p = (char*)d_ws;
  auto carve = [&](size_t bytes) {
    char* r = p;
    p += (bytes + 255) & ~(size_t)255;
    return r;
  };
  bf16_t* A1  = (bf16_t*)carve((size_t)Bn * FP * 2);
  bf16_t* Bc  = (bf16_t*)carve((size_t)CP * FP * 2);
  bf16_t* W1b = (bf16_t*)carve((size_t)Hn * CP * 2);
  bf16_t* W2b = (bf16_t*)carve((size_t)OPD * Hn * 2);
  bf16_t* H1  = (bf16_t*)carve((size_t)Bn * CP * 2);
  bf16_t* H2  = (bf16_t*)carve((size_t)Bn * Hn * 2);
  float*  x2  = (float*)carve((size_t)Bn * 4);
  float*  c2  = (float*)carve((size_t)Cn * 4);

  k_cvt_pad<<<256, 256, 0, stream>>>(inp,    Bn, Fn, A1,  Bn,  FP);
  k_cvt_pad<<<512, 256, 0, stream>>>(center, Cn, Fn, Bc,  CP,  FP);
  k_cvt_pad<<<512, 256, 0, stream>>>(w1,     Hn, Cn, W1b, Hn,  CP);
  k_cvt_pad<<<128, 256, 0, stream>>>(w2,     On, Hn, W2b, OPD, Hn);
  k_rowsumsq<<<Bn / WPB, 256, 0, stream>>>(inp,    Bn, Fn, x2);
  k_rowsumsq<<<Cn / WPB, 256, 0, stream>>>(center, Cn, Fn, c2);

  k_gemm1_rbf<<<dim3(CP / 128, Bn / 64), 256, 0, stream>>>(
      A1, Bc, H1, x2, c2, sigma, g1, bt1, mu1, vr1);
  k_gemm2<<<dim3(Hn / 128, Bn / 64), 256, 0, stream>>>(
      H1, W1b, H2, b1, g2, bt2, mu2, vr2);
  k_gemm3<<<dim3(OPD / 128, Bn / 64), 256, 0, stream>>>(H2, W2b, out, b2);
}
